// LayerDAG_37486474559636
// MI455X (gfx1250) — matrix operand order repaired and checked by reading, not compiled, not run on hardware
//
#include <hip/hip_runtime.h>
#include <math.h>

// ---------------------------------------------------------------------------
// MI455X (gfx1250): wave32 + v_wmma_f32_16x16x32_f16 GEMMs with fragment-order
// pre-packed f16 weights (global, L2-resident) and wide LDS staging for the
// activation tile. A sched_barrier pins all 8 B-fragment loads of a K-chunk
// ahead of the WMMA burst so the waits are staggered (pipelined) instead of
// loadcnt==0 stalls. Graph scatter uses f32 atomics resolving in the 192MB L2.
// ---------------------------------------------------------------------------

typedef __attribute__((ext_vector_type(16))) _Float16 v16h;
typedef __attribute__((ext_vector_type(8)))  _Float16 v8h;
typedef __attribute__((ext_vector_type(8)))  float    v8f;

#if defined(__has_builtin)
#  if __has_builtin(__builtin_amdgcn_sched_barrier)
#    define SCHED_FENCE() __builtin_amdgcn_sched_barrier(0)
#  endif
#endif
#ifndef SCHED_FENCE
#  define SCHED_FENCE() ((void)0)
#endif

__device__ __forceinline__ float gelu_exact(float x) {
  // torch F.gelu default (exact erf form)
  return 0.5f * x * (1.0f + erff(x * 0.70710678118654752440f));
}

#define TILE_M 128
#define TILE_K 32

// Pack W[K x 128] (f32, row-major) into WMMA-B fragment order (f16):
// for chunk kc = k/32, tile nt = col/16, lane = (k%32)/16*16 + col%16,
// element i = k%16:  P[((kc*8 + nt)*32 + lane)*16 + i] = W[k][col].
// Each lane's 16-half fragment is then one contiguous 32-byte load.
__global__ void pack_weights(const float* __restrict__ W,
                             _Float16* __restrict__ P, int K)
{
  int idx = blockIdx.x * blockDim.x + threadIdx.x;
  if (idx >= K * 128) return;
  int k   = idx >> 7;
  int col = idx & 127;
  int kc = k >> 5, kk = k & 31;
  int laneHi = kk >> 4, i = kk & 15;
  int nt = col >> 4, laneLo = col & 15;
  int lane = laneHi * 16 + laneLo;
  P[(((size_t)(kc * 8 + nt)) * 32 + lane) * 16 + i] = (_Float16)W[idx];
}

// C[M x 128] = act(A[M x K] @ W[K x 128] + bias).
// A: f32 row-major (ld = lda). P: f16 fragment-packed W. C: f32 (ld = ldc).
// 256 threads = 8 waves; wave w computes a 16-row x 128-col strip as 8 WMMA
// tiles. Per 32-wide K chunk: A staged to LDS as f16 (coalesced float4 global
// loads -> 2x ds_store_b128/thread), A fragment = 2x ds_load_b128, all 8 B
// fragments issued before any WMMA (sched_barrier-enforced) so waits stagger.
__global__ __launch_bounds__(256)
void gemm_bias_act(const float* __restrict__ A, int lda,
                   const _Float16* __restrict__ P,
                   const float* __restrict__ bias,
                   float* __restrict__ C, int ldc,
                   int M, int K, int act)
{
  __shared__ _Float16 sA[TILE_M * TILE_K];   // [row][k], 64 B per row

  const int tid     = threadIdx.x;
  const int lane    = tid & 31;
  const int wave    = tid >> 5;        // 0..7
  const int laneLo  = lane & 15;
  const int laneHi  = lane >> 4;       // 0 or 1
  const int rowBase = blockIdx.x * TILE_M;

  // staging assignment: unit = tid -> (row r, 16-wide k half)
  const int sr     = tid >> 1;
  const int skpart = (tid & 1) * 16;

  v8f acc[8] = {};

  for (int k0 = 0; k0 < K; k0 += TILE_K) {
    // ---- stage A tile (128 x 32) as f16, wide loads & stores -----------
    {
      int row = rowBase + sr;
      v8h h0, h1;
      if (row < M) {
        const float4* ap = (const float4*)(A + (size_t)row * lda + k0 + skpart);
        float4 f0 = ap[0], f1 = ap[1], f2 = ap[2], f3 = ap[3];
        h0[0] = (_Float16)f0.x; h0[1] = (_Float16)f0.y;
        h0[2] = (_Float16)f0.z; h0[3] = (_Float16)f0.w;
        h0[4] = (_Float16)f1.x; h0[5] = (_Float16)f1.y;
        h0[6] = (_Float16)f1.z; h0[7] = (_Float16)f1.w;
        h1[0] = (_Float16)f2.x; h1[1] = (_Float16)f2.y;
        h1[2] = (_Float16)f2.z; h1[3] = (_Float16)f2.w;
        h1[4] = (_Float16)f3.x; h1[5] = (_Float16)f3.y;
        h1[6] = (_Float16)f3.z; h1[7] = (_Float16)f3.w;
      } else {
        h0 = (v8h){}; h1 = (v8h){};
      }
      *(v8h*)&sA[sr * TILE_K + skpart]     = h0;   // ds_store_b128
      *(v8h*)&sA[sr * TILE_K + skpart + 8] = h1;   // ds_store_b128
    }
    __syncthreads();

    // ---- A fragment (contiguous k order, same as packed B): two contiguous 8-half runs -----------
    // lanes 0-15 : K runs [0,8) and [8,16); lanes 16-31: [16,24) and [24,32)
    const int arow = wave * 16 + laneLo;
    v8h alo = *(const v8h*)&sA[arow * TILE_K + laneHi * 16];        // ds_load_b128
    v8h ahi = *(const v8h*)&sA[arow * TILE_K + laneHi * 16 + 8];   // ds_load_b128
    v16h afrag = __builtin_shufflevector(alo, ahi,
        0, 1, 2, 3, 4, 5, 6, 7, 8, 9, 10, 11, 12, 13, 14, 15);

    // ---- issue ALL 8 B fragment loads, fence, then 8 WMMAs -------------
    // The sched_barrier keeps the loads from sinking to their use sites, so
    // the 8 fragments stay live in distinct VGPRs and s_wait_loadcnt values
    // stagger (14,12,...) instead of stalling at 0 before every WMMA.
    const _Float16* pbase = P + ((size_t)((k0 >> 5) * 8)) * 32 * 16;
    v16h bfr[8];
    #pragma unroll
    for (int nt = 0; nt < 8; ++nt) {
      bfr[nt] = ((const v16h*)(pbase + (size_t)nt * 32 * 16))[lane];
    }
    SCHED_FENCE();
    #pragma unroll
    for (int nt = 0; nt < 8; ++nt) {
      acc[nt] = __builtin_amdgcn_wmma_f32_16x16x32_f16(
          false, afrag, false, bfr[nt], (short)0, acc[nt], false, false);
    }
    SCHED_FENCE();
    __syncthreads();
  }

  // ---- epilogue: D layout (VGPR j -> M = j + 8*laneHi, N = laneLo) -----
  #pragma unroll
  for (int nt = 0; nt < 8; ++nt) {
    const int col = nt * 16 + laneLo;
    const float b = bias ? bias[col] : 0.0f;
    #pragma unroll
    for (int j = 0; j < 8; ++j) {
      int row = rowBase + wave * 16 + laneHi * 8 + j;
      if (row < M) {
        float v = acc[nt][j] + b;
        if (act) v = gelu_exact(v);
        C[(size_t)row * ldc + col] = v;
      }
    }
  }
}

// feat[node][0:128]   = emb_table[x_n[node]]
// feat[node][128:160] = sin(abs_level * div_term)
// feat[node][160:192] = cos(abs_level * div_term)
__global__ void build_feat(const int* __restrict__ xn,
                           const float* __restrict__ lvl,
                           const float* __restrict__ emb,
                           const float* __restrict__ divt,
                           float* __restrict__ feat, int N)
{
  int node = blockIdx.x;
  if (node >= N) return;
  int col = threadIdx.x;               // blockDim.x == 192 (6 waves)
  float v;
  if (col < 128) {
    v = emb[(size_t)xn[node] * 128 + col];
  } else {
    float a = lvl[node];
    int j = col - 128;
    v = (j < 32) ? sinf(a * divt[j]) : cosf(a * divt[j - 32]);
  }
  feat[(size_t)node * 192 + col] = v;
}

// For each edge e: acc[row[e]] += tfwd[col[e]]; acc[col[e]] += tbwd[row[e]].
// One wave per edge; each lane moves a float4 (4 atomics). acc (128 MB) is
// L2-resident on MI455X (192 MB L2), so atomics resolve in L2.
__global__ void edge_scatter(const int* __restrict__ er,
                             const int* __restrict__ ec,
                             const float* __restrict__ tf,
                             const float* __restrict__ tb,
                             float* __restrict__ acc, int E)
{
  const int lane = threadIdx.x & 31;
  int w  = (int)((blockIdx.x * blockDim.x + threadIdx.x) >> 5);
  int nw = (int)((gridDim.x * blockDim.x) >> 5);
  for (int e = w; e < E; e += nw) {
    const int r = er[e];
    const int c = ec[e];
    float4 f = reinterpret_cast<const float4*>(tf + (size_t)c * 128)[lane];
    float* ar = acc + (size_t)r * 128 + lane * 4;
    atomicAdd(ar + 0, f.x); atomicAdd(ar + 1, f.y);
    atomicAdd(ar + 2, f.z); atomicAdd(ar + 3, f.w);
    float4 g = reinterpret_cast<const float4*>(tb + (size_t)r * 128)[lane];
    float* ac = acc + (size_t)c * 128 + lane * 4;
    atomicAdd(ac + 0, g.x); atomicAdd(ac + 1, g.y);
    atomicAdd(ac + 2, g.z); atomicAdd(ac + 3, g.w);
  }
}

// hc_block[node*512 + col] = gelu(acc[node*128 + col])
__global__ void gelu_to_hc(const float* __restrict__ acc,
                           float* __restrict__ hcblk, int N)
{
  size_t idx = (size_t)blockIdx.x * blockDim.x + threadIdx.x;
  size_t total = (size_t)N * 128;
  if (idx >= total) return;
  size_t node = idx >> 7;
  int    col  = (int)(idx & 127);
  hcblk[node * 512 + col] = gelu_exact(acc[idx]);
}

extern "C" void kernel_launch(void* const* d_in, const int* in_sizes, int n_in,
                              void* d_out, int out_size, void* d_ws, size_t ws_size,
                              hipStream_t stream) {
  (void)n_in; (void)out_size; (void)ws_size;

  const int*   x_n    = (const int*)d_in[0];
  const float* abs_lv = (const float*)d_in[1];
  const int*   erow   = (const int*)d_in[2];
  const int*   ecol   = (const int*)d_in[3];
  const float* emb    = (const float*)d_in[4];
  const float* divt   = (const float*)d_in[5];
  const float* pw1    = (const float*)d_in[6];
  const float* pb1    = (const float*)d_in[7];
  const float* pw2    = (const float*)d_in[8];
  const float* pb2    = (const float*)d_in[9];
  const float* mW     = (const float*)d_in[10];
  const float* mWb    = (const float*)d_in[11];
  const float* mWt    = (const float*)d_in[12];
  const float* mWtb   = (const float*)d_in[13];
  const float* mWs    = (const float*)d_in[14];
  const float* mWsb   = (const float*)d_in[15];
  const float* ow1    = (const float*)d_in[16];
  const float* ob1    = (const float*)d_in[17];
  const float* ow2    = (const float*)d_in[18];
  const float* ob2    = (const float*)d_in[19];
  float* out = (float*)d_out;

  const int N = in_sizes[0];
  const int E = in_sizes[2];
  const int L = 3;

  // Workspace layout:
  //   hc    : N*512 f32  (concat h0..h3, ld = 512)
  //   t1    : N*128 f32
  //   tfwd  : N*128 f32  (h@W + b,  gathered by edge_col)
  //   tbwd  : N*128 f32  (h@Wt + b, gathered by edge_row)
  //   acc   : N*128 f32  (h@Ws + b, scatter-add target)
  //   wpack : 2112*128 f16 packed weights
  //   feat  : N*192 f32, aliased onto tfwd (only live before layer loop)
  char* ws = (char*)d_ws;
  float* hc   = (float*)ws;
  float* t1   = (float*)(ws + (size_t)N * 512 * sizeof(float));
  float* tfwd = t1   + (size_t)N * 128;
  float* tbwd = tfwd + (size_t)N * 128;
  float* acc  = tbwd + (size_t)N * 128;
  float* feat = tfwd;
  _Float16* wpack = (_Float16*)(acc + (size_t)N * 128);

  // packed weight sub-buffers
  size_t woff = 0;
  _Float16* Ppw1 = wpack + woff; woff += (size_t)192 * 128;
  _Float16* Ppw2 = wpack + woff; woff += (size_t)128 * 128;
  _Float16* PmW[3], *PmWt[3], *PmWs[3];
  for (int l = 0; l < L; ++l) { PmW[l]  = wpack + woff; woff += (size_t)128 * 128; }
  for (int l = 0; l < L; ++l) { PmWt[l] = wpack + woff; woff += (size_t)128 * 128; }
  for (int l = 0; l < L; ++l) { PmWs[l] = wpack + woff; woff += (size_t)128 * 128; }
  _Float16* Pow1 = wpack + woff; woff += (size_t)512 * 128;
  _Float16* Pow2 = wpack + woff; woff += (size_t)128 * 128;

  // 0) pack all weights to f16 fragment order (cheap: ~270K elements total)
  #define PACK(Wsrc, Pdst, Kdim) \
    pack_weights<<<((Kdim) * 128 + 255) / 256, 256, 0, stream>>>((Wsrc), (Pdst), (Kdim))
  PACK(pw1, Ppw1, 192);
  PACK(pw2, Ppw2, 128);
  for (int l = 0; l < L; ++l) {
    PACK(mW  + (size_t)l * 128 * 128, PmW[l],  128);
    PACK(mWt + (size_t)l * 128 * 128, PmWt[l], 128);
    PACK(mWs + (size_t)l * 128 * 128, PmWs[l], 128);
  }
  PACK(ow1, Pow1, 512);
  PACK(ow2, Pow2, 128);
  #undef PACK

  const int gblocks = (N + TILE_M - 1) / TILE_M;

  // 1) featurization: emb gather + sinusoidal PE
  build_feat<<<N, 192, 0, stream>>>(x_n, abs_lv, emb, divt, feat, N);

  // 2) input projection MLP -> hc block 0
  gemm_bias_act<<<gblocks, 256, 0, stream>>>(feat, 192, Ppw1, pb1, t1, 128, N, 192, 1);
  gemm_bias_act<<<gblocks, 256, 0, stream>>>(t1, 128, Ppw2, pb2, hc, 512, N, 128, 0);

  // 3) message-passing layers
  for (int l = 0; l < L; ++l) {
    const float* h = hc + l * 128;            // lda = 512
    gemm_bias_act<<<gblocks, 256, 0, stream>>>(h, 512, PmW[l],  mWb  + l * 128, tfwd, 128, N, 128, 0);
    gemm_bias_act<<<gblocks, 256, 0, stream>>>(h, 512, PmWt[l], mWtb + l * 128, tbwd, 128, N, 128, 0);
    gemm_bias_act<<<gblocks, 256, 0, stream>>>(h, 512, PmWs[l], mWsb + l * 128, acc,  128, N, 128, 0);
    edge_scatter<<<4096, 256, 0, stream>>>(erow, ecol, tfwd, tbwd, acc, E);
    int eb = (int)(((size_t)N * 128 + 255) / 256);
    gelu_to_hc<<<eb, 256, 0, stream>>>(acc, hc + (l + 1) * 128, N);
  }

  // 4) readout MLP over concat [N x 512]
  gemm_bias_act<<<gblocks, 256, 0, stream>>>(hc, 512, Pow1, ob1, t1, 128, N, 512, 1);
  gemm_bias_act<<<gblocks, 256, 0, stream>>>(t1, 128, Pow2, ob2, out, 128, N, 128, 0);
}
